// VGG_Cifar10_45011257262578
// MI455X (gfx1250) — 8-bit matrix operand order repaired and checked by reading, compiled, not run on hardware
//
#include <hip/hip_runtime.h>
#include <cstdint>


typedef __attribute__((ext_vector_type(8))) int   v8i;
typedef __attribute__((ext_vector_type(8))) float v8f;

__device__ __forceinline__ int8_t sgn8(float v) {
    return (v > 0.f) ? (int8_t)1 : ((v < 0.f) ? (int8_t)-1 : (int8_t)0);
}

__global__ void sgn_kernel(const float* __restrict__ src,
                           int8_t* __restrict__ dst, int n) {
    int i = blockIdx.x * blockDim.x + threadIdx.x;
    if (i < n) dst[i] = sgn8(src[i]);
}

__global__ void alpha_prep_kernel(const float* __restrict__ a,
                                  float2* __restrict__ out, int S) {
    int i = blockIdx.x * blockDim.x + threadIdx.x;
    if (i < S) out[i] = make_float2(a[i], 1.0f / a[i]);
}

__global__ void conv1_kernel(const float* __restrict__ x,
                             const float* __restrict__ w,
                             const float* __restrict__ bias,
                             const float* __restrict__ bng,
                             const float* __restrict__ bnb,
                             int8_t* __restrict__ out) {
    long idx = (long)blockIdx.x * blockDim.x + threadIdx.x;
    const long total = 64L * 34 * 34 * 128;
    if (idx >= total) return;
    int c = (int)(idx & 127);
    long t = idx >> 7;
    int xo = (int)(t % 34); t /= 34;
    int yo = (int)(t % 34);
    int b  = (int)(t / 34);
    if (xo == 0 || xo == 33 || yo == 0 || yo == 33) { out[idx] = 0; return; }
    int X = xo - 1, Y = yo - 1;
    float acc = bias[c];
    #pragma unroll
    for (int ic = 0; ic < 3; ++ic)
        #pragma unroll
        for (int ky = 0; ky < 3; ++ky) {
            int iy = Y + ky - 1;
            if (iy < 0 || iy >= 32) continue;
            #pragma unroll
            for (int kx = 0; kx < 3; ++kx) {
                int ix = X + kx - 1;
                if (ix < 0 || ix >= 32) continue;
                float wv = w[((c * 3 + ic) * 3 + ky) * 3 + kx];
                float sw = (wv > 0.f) ? 1.f : ((wv < 0.f) ? -1.f : 0.f);
                acc += sw * x[((b * 3 + ic) * 32 + iy) * 32 + ix];
            }
        }
    out[idx] = sgn8(acc * bng[c] + bnb[c]);
}

#define PSUM_NT 4
__global__ __launch_bounds__(128)
void psum_stage_kernel(const int8_t* __restrict__ xbin,
                       const int8_t* __restrict__ wbin,
                       const float*  __restrict__ bias,
                       const float2* __restrict__ alpha_ai,
                       float*        __restrict__ hout,
                       int H, int W, int Co, int S, int Cin) {
    const int lane = threadIdx.x & 31;
    const int wave = threadIdx.x >> 5;
    const int col  = lane & 15;
    const int hi   = lane >> 4;

    const int octile = blockIdx.x;
    const int ntbase = (blockIdx.y * 4 + wave) * PSUM_NT;
    const int HW   = H * W;
    const int Npos = 64 * HW;
    if (ntbase * 16 >= Npos) return;

    const int Wp = W + 2, Hp = H + 2;

    int xoff[PSUM_NT];
    int poff[PSUM_NT];
    #pragma unroll
    for (int t = 0; t < PSUM_NT; ++t) {
        int n   = (ntbase + t) * 16 + col;
        int b   = n / HW;
        int rem = n - b * HW;
        int y   = rem / W;
        int x   = rem - y * W;
        xoff[t] = ((b * Hp + y) * Wp + x) * Cin;
        poff[t] = b * Co * HW + rem;
    }

    int akoff[8], bkoff[8];
    #pragma unroll
    for (int v = 0; v < 8; ++v) {
        akoff[v] = ((v >> 2) * 32) + (hi * 16) + ((v & 3) * 4);
        bkoff[v] = ((v >> 2) * 32) + (hi * 16) + ((v & 3) * 4);
    }
    const int m_lane = octile * 16 + col;

    v8f acc[PSUM_NT] = {};

    for (int s = 0; s < S; ++s) {
        int g  = s / 9;
        int r9 = s - g * 9;
        int dj = r9 / 3;
        int di = r9 - dj * 3;
        const int shift = (di * Wp + dj) * Cin + g * 128;
        const int8_t* wp = wbin + ((long)s * Co + m_lane) * 128;

        v8i a0, a1;
        #pragma unroll
        for (int v = 0; v < 8; ++v) {
            a0[v] = *(const int*)(wp + akoff[v]);
            a1[v] = *(const int*)(wp + akoff[v] + 64);
        }

        const float2 ai = alpha_ai[s];
        const float* bp = bias + (long)s * Co + octile * 16 + hi * 8;
        v8f bv;
        #pragma unroll
        for (int r = 0; r < 8; ++r) bv[r] = bp[r];

        #pragma unroll
        for (int t = 0; t < PSUM_NT; ++t) {
            const int8_t* xp = xbin + xoff[t] + shift;
            v8i b0, b1;
            #pragma unroll
            for (int v = 0; v < 8; ++v) {
                b0[v] = *(const int*)(xp + bkoff[v]);
                b1[v] = *(const int*)(xp + bkoff[v] + 64);
            }
            v8i c = {};
            c = __builtin_amdgcn_wmma_i32_16x16x64_iu8(true, a0, true, b0, c, false, false);
            c = __builtin_amdgcn_wmma_i32_16x16x64_iu8(true, a1, true, b1, c, false, false);

            #pragma unroll
            for (int r = 0; r < 8; ++r) {
                float yv = (float)c[r] + bv[r];
                float yc = fminf(fmaxf(yv * ai.y, -4.0f), 3.0f);
                acc[t][r] += rintf(yc) * ai.x;
            }
        }
    }

    #pragma unroll
    for (int t = 0; t < PSUM_NT; ++t)
        #pragma unroll
        for (int r = 0; r < 8; ++r) {
            int oc = octile * 16 + r + 8 * hi;
            hout[(long)poff[t] + (long)oc * HW] = acc[t][r];
        }
}

__global__ void pool_bn_bin_kernel(const float* __restrict__ h,
                                   const float* __restrict__ g,
                                   const float* __restrict__ bb,
                                   int8_t* __restrict__ out,
                                   int C, int H, int W, int pool) {
    int Ho = pool ? H / 2 : H, Wo = pool ? W / 2 : W;
    long total = 64L * (Ho + 2) * (Wo + 2) * C;
    long idx = (long)blockIdx.x * blockDim.x + threadIdx.x;
    if (idx >= total) return;
    int c = (int)(idx % C);
    long t = idx / C;
    int xo = (int)(t % (Wo + 2)); t /= (Wo + 2);
    int yo = (int)(t % (Ho + 2));
    int b  = (int)(t / (Ho + 2));
    if (xo == 0 || xo == Wo + 1 || yo == 0 || yo == Ho + 1) { out[idx] = 0; return; }
    int Y = yo - 1, X = xo - 1;
    float v;
    if (pool) {
        const float* p = h + ((long)(b * C + c) * H + 2 * Y) * W + 2 * X;
        v = fmaxf(fmaxf(p[0], p[1]), fmaxf(p[W], p[W + 1]));
    } else {
        v = h[((long)(b * C + c) * H + Y) * W + X];
    }
    out[idx] = sgn8(v * g[c] + bb[c]);
}

__global__ void pool_bn_bin_flat_kernel(const float* __restrict__ h,
                                        const float* __restrict__ g,
                                        const float* __restrict__ bb,
                                        int8_t* __restrict__ out) {
    int idx = blockIdx.x * blockDim.x + threadIdx.x;
    if (idx >= 64 * 512 * 16) return;
    int p = idx & 15;
    int c = (idx >> 4) & 511;
    int b = idx >> 13;
    int Y = p >> 2, X = p & 3;
    const float* ph = h + ((long)(b * 512 + c) * 8 + 2 * Y) * 8 + 2 * X;
    float v = fmaxf(fmaxf(ph[0], ph[1]), fmaxf(ph[8], ph[9]));
    out[(long)b * 8192 + c * 16 + p] = sgn8(v * g[c] + bb[c]);
}

__global__ __launch_bounds__(32)
void fc_bin_kernel(const int8_t* __restrict__ A,
                   const int8_t* __restrict__ Wt,
                   const float*  __restrict__ bias,
                   const float*  __restrict__ g,
                   const float*  __restrict__ bb,
                   int8_t* __restrict__ out,
                   int K, int N) {
    const int lane = threadIdx.x & 31;
    const int col  = lane & 15;
    const int hi   = lane >> 4;
    const int ntile = blockIdx.x;
    const int mtile = blockIdx.y;

    int akoff[8], bkoff[8];
    #pragma unroll
    for (int v = 0; v < 8; ++v) {
        akoff[v] = ((v >> 2) * 32) + (hi * 16) + ((v & 3) * 4);
        bkoff[v] = ((v >> 2) * 32) + (hi * 16) + ((v & 3) * 4);
    }
    const int8_t* arow = A  + (long)(mtile * 16 + col) * K;
    const int8_t* brow = Wt + (long)(ntile * 16 + col) * K;

    v8i c = {};
    for (int k0 = 0; k0 < K; k0 += 64) {
        v8i a, b;
        #pragma unroll
        for (int v = 0; v < 8; ++v) {
            a[v] = *(const int*)(arow + k0 + akoff[v]);
            b[v] = *(const int*)(brow + k0 + bkoff[v]);
        }
        c = __builtin_amdgcn_wmma_i32_16x16x64_iu8(true, a, true, b, c, false, false);
    }
    #pragma unroll
    for (int r = 0; r < 8; ++r) {
        int m = mtile * 16 + r + 8 * hi;
        int n = ntile * 16 + col;
        float yv = (float)c[r] + bias[n];
        out[(long)m * N + n] = sgn8(yv * g[n] + bb[n]);
    }
}

__global__ void fc3_kernel(const int8_t* __restrict__ f,
                           const int8_t* __restrict__ w,
                           const float* __restrict__ bias,
                           const float* __restrict__ g,
                           const float* __restrict__ bb,
                           float* __restrict__ outp) {
    int idx = blockIdx.x * blockDim.x + threadIdx.x;
    if (idx >= 640) return;
    int o = idx % 10, b = idx / 10;
    const int8_t* fp = f + b * 1024;
    const int8_t* wp = w + o * 1024;
    int acc = 0;
    for (int k = 0; k < 1024; ++k) acc += (int)fp[k] * (int)wp[k];
    float yv = (float)acc + bias[o];
    outp[idx] = yv * g[o] + bb[o];
}

extern "C" void kernel_launch(void* const* d_in, const int* in_sizes, int n_in,
                              void* d_out, int out_size, void* d_ws, size_t ws_size,
                              hipStream_t stream) {
    const float* x       = (const float*)d_in[0];
    const float* conv1_w = (const float*)d_in[1];
    const float* conv1_b = (const float*)d_in[2];
    const float* sw[5]   = {(const float*)d_in[3],  (const float*)d_in[6],
                            (const float*)d_in[9],  (const float*)d_in[12],
                            (const float*)d_in[15]};
    const float* sb[5]   = {(const float*)d_in[4],  (const float*)d_in[7],
                            (const float*)d_in[10], (const float*)d_in[13],
                            (const float*)d_in[16]};
    const float* sa[5]   = {(const float*)d_in[5],  (const float*)d_in[8],
                            (const float*)d_in[11], (const float*)d_in[14],
                            (const float*)d_in[17]};
    const float* bng[6]  = {(const float*)d_in[18], (const float*)d_in[20],
                            (const float*)d_in[22], (const float*)d_in[24],
                            (const float*)d_in[26], (const float*)d_in[28]};
    const float* bnb[6]  = {(const float*)d_in[19], (const float*)d_in[21],
                            (const float*)d_in[23], (const float*)d_in[25],
                            (const float*)d_in[27], (const float*)d_in[29]};
    const float* fc1_w = (const float*)d_in[30]; const float* fc1_b = (const float*)d_in[31];
    const float* fc2_w = (const float*)d_in[32]; const float* fc2_b = (const float*)d_in[33];
    const float* fc3_w = (const float*)d_in[34]; const float* fc3_b = (const float*)d_in[35];
    const float* fbn1g = (const float*)d_in[36]; const float* fbn1b = (const float*)d_in[37];
    const float* fbn2g = (const float*)d_in[38]; const float* fbn2b = (const float*)d_in[39];
    const float* fbn3g = (const float*)d_in[40]; const float* fbn3b = (const float*)d_in[41];
    float* out = (float*)d_out;

    int8_t* ws = (int8_t*)d_ws;
    const size_t OFF_XBIN = 0;
    const size_t OFF_H    = 10u << 20;
    const size_t OFF_W    = OFF_H + (34u << 20);
    const size_t WSZ[5]   = {9u*128*128, 9u*256*128, 18u*256*128, 18u*512*128, 36u*512*128};
    const int    SN[5]    = {9, 9, 18, 18, 36};
    size_t o_w[5]; size_t cur = OFF_W;
    for (int i = 0; i < 5; ++i) { o_w[i] = cur; cur += WSZ[i]; }
    const size_t o_fc1w  = cur;           cur += 1024u * 8192;
    const size_t o_fc2w  = cur;           cur += 1024u * 1024;
    const size_t o_fc3w  = cur;           cur += 10u * 1024 + 256;
    const size_t o_fbin  = cur;           cur += 64u * 8192;
    const size_t o_f1    = cur;           cur += 64u * 1024;
    const size_t o_f2    = cur;           cur += 64u * 1024;
    const size_t o_alpha = cur;

    int8_t* xbin = ws + OFF_XBIN;
    float*  hbuf = (float*)(ws + OFF_H);

    auto sgn_launch = [&](const float* src, int8_t* dst, int n) {
        sgn_kernel<<<(n + 255) / 256, 256, 0, stream>>>(src, dst, n);
    };
    for (int i = 0; i < 5; ++i) {
        sgn_launch(sw[i], ws + o_w[i], (int)WSZ[i]);
        alpha_prep_kernel<<<1, 64, 0, stream>>>(
            sa[i], (float2*)(ws + o_alpha + (size_t)i * 36 * sizeof(float2)), SN[i]);
    }
    sgn_launch(fc1_w, ws + o_fc1w, 1024 * 8192);
    sgn_launch(fc2_w, ws + o_fc2w, 1024 * 1024);
    sgn_launch(fc3_w, ws + o_fc3w, 10 * 1024);

    {
        long total = 64L * 34 * 34 * 128;
        conv1_kernel<<<(unsigned)((total + 255) / 256), 256, 0, stream>>>(
            x, conv1_w, conv1_b, bng[0], bnb[0], xbin);
    }

    struct Stage { int H, W, Co, S, Cin, bn_idx, pool; };
    const Stage st[5] = {
        {32, 32, 128,  9, 128, 1, 1},
        {16, 16, 256,  9, 128, 2, 0},
        {16, 16, 256, 18, 256, 3, 1},
        { 8,  8, 512, 18, 256, 4, 0},
        { 8,  8, 512, 36, 512, 5, 1},
    };

    for (int i = 0; i < 5; ++i) {
        const Stage& s = st[i];
        int Npos = 64 * s.H * s.W;
        dim3 grid(s.Co / 16, Npos / (64 * PSUM_NT));
        psum_stage_kernel<<<grid, 128, 0, stream>>>(
            xbin, ws + o_w[i], sb[i],
            (const float2*)(ws + o_alpha + (size_t)i * 36 * sizeof(float2)),
            hbuf, s.H, s.W, s.Co, s.S, s.Cin);

        if (i < 4) {
            int Ho = s.pool ? s.H / 2 : s.H, Wo = s.pool ? s.W / 2 : s.W;
            long total = 64L * (Ho + 2) * (Wo + 2) * s.Co;
            pool_bn_bin_kernel<<<(unsigned)((total + 255) / 256), 256, 0, stream>>>(
                hbuf, bng[s.bn_idx], bnb[s.bn_idx], xbin, s.Co, s.H, s.W, s.pool);
        } else {
            pool_bn_bin_flat_kernel<<<(64 * 512 * 16) / 256, 256, 0, stream>>>(
                hbuf, bng[5], bnb[5], ws + o_fbin);
        }
    }

    fc_bin_kernel<<<dim3(1024 / 16, 4), 32, 0, stream>>>(
        ws + o_fbin, ws + o_fc1w, fc1_b, fbn1g, fbn1b, ws + o_f1, 8192, 1024);
    fc_bin_kernel<<<dim3(1024 / 16, 4), 32, 0, stream>>>(
        ws + o_f1, ws + o_fc2w, fc2_b, fbn2g, fbn2b, ws + o_f2, 1024, 1024);
    fc3_kernel<<<(640 + 255) / 256, 256, 0, stream>>>(
        ws + o_f2, ws + o_fc3w, fc3_b, fbn3g, fbn3b, out);
    (void)in_sizes; (void)n_in; (void)out_size; (void)ws_size;
}
